// EdgeRelativeEmbedMLP_22651657519356
// MI455X (gfx1250) — matrix operand order repaired and checked by reading, not compiled, not run on hardware
//
#include <hip/hip_runtime.h>

#define ADIM 128
#define VDIM 64
#define TILE 128      // edges per block
#define THREADS 256   // 8 waves (wave32)

// f16 weight image in d_ws: W1 padded [128][32] | W2 [128][128] | W3 [128][128]
#define W1H_ELEMS (ADIM * 32)
#define W23_ELEMS (ADIM * ADIM)
#define WSH_ELEMS (W1H_ELEMS + 2 * W23_ELEMS)   // 36864 f16 = 73728 bytes

typedef __attribute__((ext_vector_type(16))) _Float16 v16h;
typedef __attribute__((ext_vector_type(8)))  float    v8f;
typedef __attribute__((ext_vector_type(4)))  unsigned tdm_u32x4;
typedef __attribute__((ext_vector_type(8)))  int      tdm_i32x8;
typedef __attribute__((ext_vector_type(4)))  int      tdm_i32x4;

union ABPack { v16h v; unsigned u[8]; };

// A-matrix (16xK f16) fragment load from row-major LDS [m][k] (stride elems).
// ISA 7.12.2: lanes 0-15 hold K={0..7,16..23}, lanes 16-31 K={8..15,24..31}.
__device__ __forceinline__ v16h load_A16(const _Float16* base, int stride, int m, int sel, int kc) {
  ABPack r;
#pragma unroll
  for (int j = 0; j < 8; ++j) {
    int kk = (j < 4) ? (sel * 8 + 2 * j) : (16 + sel * 8 + 2 * (j - 4));
    r.u[j] = *(const unsigned*)(base + m * stride + kc + kk);
  }
  return r.v;
}

// B-matrix (Kx16 f16) fragment: column n per lane; lanes 0-15 hold K={0..7,16..23} of the
// chunk, lanes 16-31 hold K={8..15,24..31}. LDS holds W in [n][k] row-major (B[k][n]=W[n][k]).
__device__ __forceinline__ v16h load_B16(const _Float16* base, int stride, int n, int sel, int kc) {
  ABPack r;
#pragma unroll
  for (int j = 0; j < 8; ++j) {
    int kk = (j < 4) ? (sel * 8 + 2 * j) : (16 + sel * 8 + 2 * (j - 4));
    r.u[j] = *(const unsigned*)(base + n * stride + kc + kk);
  }
  return r.v;
}

#if defined(__HIP_DEVICE_COMPILE__) && __has_builtin(__builtin_amdgcn_tensor_load_to_lds)
#define HAVE_TDM 1
#else
#define HAVE_TDM 0
#endif

#if HAVE_TDM
// Issue one TDM descriptor: flat 1-D copy of `nelem` f16 from global to LDS.
// D# layout per CDNA5 ISA ch.8: group0 = {count/type/addr}, group1 = {dims}.
__device__ __forceinline__ void tdm_load_1d_f16(unsigned lds_byte_addr,
                                                const void* gptr, unsigned nelem) {
  unsigned long long ga = (unsigned long long)(uintptr_t)gptr;
  tdm_u32x4 g0;
  g0[0] = 1u;                                            // count=1 valid descriptor
  g0[1] = lds_byte_addr;                                 // lds_addr[31:0]
  g0[2] = (unsigned)(ga & 0xFFFFFFFFu);                  // global_addr[31:0]
  g0[3] = (unsigned)((ga >> 32) & 0x01FFFFFFu)           // global_addr[56:32]
        | (2u << 30);                                    // type=2 ("image")
  tdm_i32x8 g1;
  g1[0] = (int)(1u << 16);                               // data_size=1 (2 bytes), mask=0
  g1[1] = (int)((nelem & 0xFFFFu) << 16);                // tensor_dim0[15:0] @ bits 63:48
  g1[2] = (int)(((nelem >> 16) & 0xFFFFu)                // tensor_dim0[31:16] @ bits 79:64
        | (1u << 16));                                   // tensor_dim1=1 @ bits 111:80
  g1[3] = (int)((nelem & 0xFFFFu) << 16);                // tile_dim0 @ bits 127:112
  g1[4] = 0;                                             // tile_dim1=0, tile_dim2=0
  g1[5] = (int)nelem;                                    // tensor_dim0_stride[31:0]
  g1[6] = 0;
  g1[7] = 0;
  tdm_i32x4 gz = {0, 0, 0, 0};
#if defined(__clang_major__) && __clang_major__ >= 23
  tdm_i32x8 gz8 = {0, 0, 0, 0, 0, 0, 0, 0};
  __builtin_amdgcn_tensor_load_to_lds(g0, g1, gz, gz, gz8, 0);
#else
  __builtin_amdgcn_tensor_load_to_lds(g0, g1, gz, gz, 0);
#endif
}
#endif

__device__ __forceinline__ unsigned lds_byte_addr(const void* p) {
  // Generic LDS addresses carry the wave-relative LDS offset in the low 32 bits.
  return (unsigned)(uintptr_t)p;
}

template <bool USE_TDM>
__global__ __launch_bounds__(THREADS, 1)
void edge_mlp_kernel(const float* __restrict__ pos0, const float* __restrict__ pos1,
                     const long long* __restrict__ src, const long long* __restrict__ dst,
                     const float* __restrict__ Wv,
                     const float* __restrict__ W1, const float* __restrict__ b1,
                     const float* __restrict__ W2, const float* __restrict__ b2,
                     const float* __restrict__ W3, const float* __restrict__ b3,
                     const _Float16* __restrict__ Wh,   // prepacked f16 weights (d_ws)
                     float* __restrict__ aout, float* __restrict__ vout,
                     int N, int E)
{
  __shared__ __align__(16) _Float16 W1B[W1H_ELEMS];   // 8 KB  (K padded 6->32, zeros)
  __shared__ __align__(16) _Float16 W2B[W23_ELEMS];   // 32 KB
  __shared__ __align__(16) _Float16 W3B[W23_ELEMS];   // 32 KB
  __shared__ _Float16 normsA[TILE * 32];              // 8 KB  (layer-1 A, zero padded)
  __shared__ _Float16 hA[TILE * ADIM];                // 32 KB
  __shared__ _Float16 hB[TILE * ADIM];                // 32 KB
  __shared__ float    vecsS[TILE * 18];               // 9 KB
  __shared__ float    biasS[3 * ADIM];                // 1.5 KB
  __shared__ float    WvS[VDIM * 6];                  // 1.5 KB

  const int tid = threadIdx.x;
  const int b = blockIdx.y;
  const size_t ebase = (size_t)blockIdx.x * TILE;

  // ---- Phase A: stage weights to LDS ----
  if (USE_TDM) {
#if HAVE_TDM
    if (tid < 32) {   // wave 0 issues the async tensor DMA; overlaps with geometry
      tdm_load_1d_f16(lds_byte_addr(W1B), Wh, W1H_ELEMS);
      tdm_load_1d_f16(lds_byte_addr(W2B), Wh + W1H_ELEMS, W23_ELEMS);
      tdm_load_1d_f16(lds_byte_addr(W3B), Wh + W1H_ELEMS + W23_ELEMS, W23_ELEMS);
    }
#else
    // No TDM builtin on this toolchain: plain cooperative f16 copy from d_ws.
    for (int idx = tid; idx < W1H_ELEMS; idx += THREADS) W1B[idx] = Wh[idx];
    for (int idx = tid; idx < W23_ELEMS; idx += THREADS) {
      W2B[idx] = Wh[W1H_ELEMS + idx];
      W3B[idx] = Wh[W1H_ELEMS + W23_ELEMS + idx];
    }
#endif
  } else {
    // Fallback: convert f32 weights in-kernel.
    for (int idx = tid; idx < W1H_ELEMS; idx += THREADS) {
      int n = idx >> 5, k = idx & 31;
      W1B[idx] = (k < 6) ? (_Float16)W1[n * 6 + k] : (_Float16)0.f;
    }
    for (int idx = tid; idx < W23_ELEMS; idx += THREADS) {
      W2B[idx] = (_Float16)W2[idx];
      W3B[idx] = (_Float16)W3[idx];
    }
  }
  if (tid < ADIM) {
    biasS[tid]            = b1[tid];
    biasS[ADIM + tid]     = b2[tid];
    biasS[2 * ADIM + tid] = b3[tid];
  }
  for (int idx = tid; idx < VDIM * 6; idx += THREADS) WvS[idx] = Wv[idx];

  // ---- Phase B: per-edge geometry (gather, 6 vectors, 6 norms) ----
  if (tid < TILE) {
#pragma unroll
    for (int k = 0; k < 32; ++k) normsA[tid * 32 + k] = (_Float16)0.f;

    size_t e = ebase + (size_t)tid;
    if (e < (size_t)E) {
      long long s = src[e];
      long long d = dst[e];
      const float* P0 = pos0 + (size_t)b * N * 3;
      const float* P1 = pos1 + (size_t)b * N * 3;
      float p0s[3], p0d[3], p1s[3], p1d[3];
#pragma unroll
      for (int k = 0; k < 3; ++k) {
        p0s[k] = P0[(size_t)s * 3 + k];
        p0d[k] = P0[(size_t)d * 3 + k];
        p1s[k] = P1[(size_t)s * 3 + k];
        p1d[k] = P1[(size_t)d * 3 + k];
      }
      float vv[6][3];
#pragma unroll
      for (int k = 0; k < 3; ++k) {
        vv[0][k] = p0d[k] - p0s[k];
        vv[1][k] = p1d[k] - p1s[k];
        vv[2][k] = p1s[k] - p0s[k];
        vv[3][k] = p1d[k] - p0d[k];
        vv[4][k] = p1s[k] - p0d[k];
        vv[5][k] = p1d[k] - p0s[k];
      }
#pragma unroll
      for (int i = 0; i < 6; ++i) {
        float nrm = sqrtf(vv[i][0] * vv[i][0] + vv[i][1] * vv[i][1] + vv[i][2] * vv[i][2]);
        normsA[tid * 32 + i] = (_Float16)nrm;
#pragma unroll
        for (int k = 0; k < 3; ++k) vecsS[tid * 18 + i * 3 + k] = vv[i][k];
      }
    }
  }
#if HAVE_TDM
  if (USE_TDM && tid < 32) __builtin_amdgcn_s_wait_tensorcnt(0);  // wave 0: DMA done
#endif
  __syncthreads();

  // ---- Phase C: v_out = (Wv @ vecs)/3, coalesced cooperative stores ----
  {
    float* vbase = vout + ((size_t)b * E + ebase) * (VDIM * 3);
    for (int idx = tid; idx < TILE * VDIM * 3; idx += THREADS) {
      int el = idx / (VDIM * 3);
      int j  = idx - el * (VDIM * 3);
      int o  = j / 3;
      int kk = j - o * 3;
      float s = 0.f;
#pragma unroll
      for (int i = 0; i < 6; ++i) s += WvS[o * 6 + i] * vecsS[el * 18 + i * 3 + kk];
      if (ebase + (size_t)el < (size_t)E) vbase[(size_t)el * (VDIM * 3) + j] = s * (1.f / 3.f);
    }
  }

  // ---- WMMA phases: 8 waves, each owns a 16-edge stripe x full 128 outputs ----
  const int wave = tid >> 5;
  const int lane = tid & 31;
  const int sel  = lane >> 4;
  const int ml   = lane & 15;
  const int nl   = lane & 15;
  const int mbase = wave * 16;

  // Layer 1: [128x32(pad)] x [32x128] -> hA, leaky
  {
    v16h a = load_A16(normsA, 32, mbase + ml, sel, 0);
#pragma unroll
    for (int nt = 0; nt < 8; ++nt) {
      int ncol = nt * 16 + nl;
      v16h bb = load_B16(W1B, 32, ncol, sel, 0);
      v8f c = {};
      c = __builtin_amdgcn_wmma_f32_16x16x32_f16(false, a, false, bb, (short)0, c, false, false);
      float bias = biasS[ncol];
#pragma unroll
      for (int r = 0; r < 8; ++r) {
        float x = c[r] + bias;
        x = (x >= 0.f) ? x : 0.2f * x;
        hA[(mbase + r + 8 * sel) * ADIM + ncol] = (_Float16)x;
      }
    }
  }
  __syncthreads();

  // Layer 2: [128x128] x [128x128] -> hB, leaky
  {
    v16h a4[4];
#pragma unroll
    for (int kc = 0; kc < 4; ++kc) a4[kc] = load_A16(hA, ADIM, mbase + ml, sel, kc * 32);
#pragma unroll
    for (int nt = 0; nt < 8; ++nt) {
      int ncol = nt * 16 + nl;
      v8f c = {};
#pragma unroll
      for (int kc = 0; kc < 4; ++kc) {
        v16h bb = load_B16(W2B, ADIM, ncol, sel, kc * 32);
        c = __builtin_amdgcn_wmma_f32_16x16x32_f16(false, a4[kc], false, bb, (short)0, c, false, false);
      }
      float bias = biasS[ADIM + ncol];
#pragma unroll
      for (int r = 0; r < 8; ++r) {
        float x = c[r] + bias;
        x = (x >= 0.f) ? x : 0.2f * x;
        hB[(mbase + r + 8 * sel) * ADIM + ncol] = (_Float16)x;
      }
    }
  }
  __syncthreads();

  // Layer 3: [128x128] x [128x128] -> a_out (f32, +bias, no activation)
  {
    v16h a4[4];
#pragma unroll
    for (int kc = 0; kc < 4; ++kc) a4[kc] = load_A16(hB, ADIM, mbase + ml, sel, kc * 32);
#pragma unroll
    for (int nt = 0; nt < 8; ++nt) {
      int ncol = nt * 16 + nl;
      v8f c = {};
#pragma unroll
      for (int kc = 0; kc < 4; ++kc) {
        v16h bb = load_B16(W3B, ADIM, ncol, sel, kc * 32);
        c = __builtin_amdgcn_wmma_f32_16x16x32_f16(false, a4[kc], false, bb, (short)0, c, false, false);
      }
      float bias = biasS[2 * ADIM + ncol];
#pragma unroll
      for (int r = 0; r < 8; ++r) {
        size_t e = ebase + (size_t)(mbase + r + 8 * sel);
        if (e < (size_t)E)
          aout[((size_t)b * E + e) * ADIM + ncol] = c[r] + bias;
      }
    }
  }
}

// One-shot weight pack: f32 -> f16 image in d_ws (run each launch; deterministic).
__global__ void prep_weights_kernel(const float* __restrict__ W1,
                                    const float* __restrict__ W2,
                                    const float* __restrict__ W3,
                                    _Float16* __restrict__ Wh)
{
  int i = blockIdx.x * blockDim.x + threadIdx.x;
  if (i < W1H_ELEMS) {
    int n = i >> 5, k = i & 31;
    Wh[i] = (k < 6) ? (_Float16)W1[n * 6 + k] : (_Float16)0.f;
  } else if (i < W1H_ELEMS + W23_ELEMS) {
    Wh[i] = (_Float16)W2[i - W1H_ELEMS];
  } else if (i < WSH_ELEMS) {
    Wh[i] = (_Float16)W3[i - (W1H_ELEMS + W23_ELEMS)];
  }
}

extern "C" void kernel_launch(void* const* d_in, const int* in_sizes, int n_in,
                              void* d_out, int out_size, void* d_ws, size_t ws_size,
                              hipStream_t stream) {
  const float*     pos0 = (const float*)d_in[0];
  const float*     pos1 = (const float*)d_in[1];
  const long long* src  = (const long long*)d_in[2];
  const long long* dst  = (const long long*)d_in[3];
  const float*     Wv   = (const float*)d_in[4];
  const float*     W1   = (const float*)d_in[5];
  const float*     b1   = (const float*)d_in[6];
  const float*     W2   = (const float*)d_in[7];
  const float*     b2   = (const float*)d_in[8];
  const float*     W3   = (const float*)d_in[9];
  const float*     b3   = (const float*)d_in[10];

  const int B = 4;
  const int N = in_sizes[0] / (B * 3);
  const int E = in_sizes[2];

  float* aout = (float*)d_out;
  float* vout = aout + (size_t)B * E * ADIM;

  dim3 grid((E + TILE - 1) / TILE, B);
  const size_t ws_need = (size_t)WSH_ELEMS * 2;

  if (ws_size >= ws_need && d_ws != nullptr) {
    _Float16* Wh = (_Float16*)d_ws;
    prep_weights_kernel<<<(WSH_ELEMS + 255) / 256, 256, 0, stream>>>(W1, W2, W3, Wh);
    edge_mlp_kernel<true><<<grid, THREADS, 0, stream>>>(pos0, pos1, src, dst, Wv,
                                                        W1, b1, W2, b2, W3, b3, Wh,
                                                        aout, vout, N, E);
  } else {
    edge_mlp_kernel<false><<<grid, THREADS, 0, stream>>>(pos0, pos1, src, dst, Wv,
                                                         W1, b1, W2, b2, W3, b3, nullptr,
                                                         aout, vout, N, E);
  }
}
